// GaussianBlur_38963943309988
// MI455X (gfx1250) — matrix operand order repaired and checked by reading, not compiled, not run on hardware
//
#include <hip/hip_runtime.h>

typedef __attribute__((ext_vector_type(16))) _Float16 v16h;
typedef __attribute__((ext_vector_type(8)))  float    v8f;

#define W 512
#define RAD 8

// normalized Gaussian taps: cv2.getGaussianKernel(17, 2.0), sums to 1
__device__ static const float GTAP[17] = {
  6.691e-05f,    4.3635e-04f,   2.21596e-03f,  8.76431e-03f,
  2.699596e-02f, 6.475994e-02f, 1.2098749e-01f, 1.7603573e-01f,
  1.9947466e-01f,
  1.7603573e-01f, 1.2098749e-01f, 6.475994e-02f, 2.699596e-02f,
  8.76431e-03f,  2.21596e-03f,  4.3635e-04f,   6.691e-05f
};

__device__ __forceinline__ int refl(int i) {            // reflect_101 for N=512
  i = (i < 0) ? -i : i;
  return (i > W - 1) ? (2 * (W - 1) - i) : i;
}

__global__ __launch_bounds__(256)
void gauss17_wmma(const float* __restrict__ in, float* __restrict__ out) {
  // per-wave staging: 32x32 input patch (f16) and 32x16 intermediate (f16)
  __shared__ _Float16 patch[8][32][40];   // row stride 40 halves to stagger banks
  __shared__ _Float16 mid  [8][32][18];
  __shared__ _Float16 gtab [64];          // zero-padded taps: gtab[16+d]=g[d], d in [0,16]

  const int lane  = threadIdx.x & 31;
  const int wv    = threadIdx.x >> 5;
  const int wgid  = blockIdx.x * 8 + wv;          // one 16x16 tile per wave
  const int plane = wgid >> 10;                   // 96 planes (B*C)
  const int t     = wgid & 1023;
  const int ty    = t >> 5, tx = t & 31;
  const int row0  = ty * 16, col0 = tx * 16;

  const float* ip = in  + ((size_t)plane << 18);
  float*       op = out + ((size_t)plane << 18);

  // ---- one-time per block: stage zero-padded f16 tap table
  if (threadIdx.x < 64) {
    const int i = threadIdx.x;
    const float v = (i >= 16 && i <= 32) ? GTAP[i - 16] : 0.0f;
    gtab[i] = (_Float16)v;
  }

  // ---- stage input rows row0-8..row0+23, cols col0-8..col0+23 into LDS (f16)
  const bool edge = (tx == 0) || (tx == 31) || (ty == 0) || (ty == 31);
  if (!edge) {
    const float* src = ip + (size_t)(row0 - RAD + lane) * W + (col0 - RAD); // 16B aligned
    #pragma unroll
    for (int i = 0; i < 8; ++i) {
      float4 f = ((const float4*)src)[i];
      patch[wv][lane][4*i+0] = (_Float16)f.x;
      patch[wv][lane][4*i+1] = (_Float16)f.y;
      patch[wv][lane][4*i+2] = (_Float16)f.z;
      patch[wv][lane][4*i+3] = (_Float16)f.w;
    }
  } else {
    const int gr = refl(row0 - RAD + lane);
    #pragma unroll 4
    for (int c = 0; c < 32; ++c) {
      const int gc = refl(col0 - RAD + c);
      patch[wv][lane][c] = (_Float16)ip[(size_t)gr * W + gc];
    }
  }
  __syncthreads();   // covers both patch and gtab

  const int q  = lane & 15;   // M-row for A operands / N-col for B,C,D operands
  const int hh = lane >> 4;   // lane half

  // banded kernel as B operand (32x16): Bw[k][n] = g[k-n]
  // B layout: element e -> K = (e<8 ? hh*8+e : 16 + hh*8 + (e-8))
  v16h Bw;
  #pragma unroll
  for (int e = 0; e < 16; ++e)
    Bw[e] = gtab[16 + ((e < 8) ? (hh * 8 + e) : (16 + hh * 8 + (e - 8))) - q];           // zero outside band

  // banded kernel as A operand (16x32): A2[m][k] = g[k-m]
  // A layout: element e -> K = (e<8 ? hh*8+e : 16 + hh*8 + (e-8))
  v16h A2;
  #pragma unroll
  for (int e = 0; e < 16; ++e) {
    const int K = (e < 8) ? (hh * 8 + e) : (16 + hh * 8 + (e - 8));
    A2[e] = gtab[16 + K - q];
  }

  // ---- horizontal pass: two 16x32 input tiles x banded B -> 32x16 intermediate
  v16h a0, a1;
  #pragma unroll
  for (int e = 0; e < 16; ++e) {
    const int K = (e < 8) ? (hh * 8 + e) : (16 + hh * 8 + (e - 8));
    a0[e] = patch[wv][q][K];
    a1[e] = patch[wv][16 + q][K];
  }
  v8f c0 = {}, c1 = {};
  c0 = __builtin_amdgcn_wmma_f32_16x16x32_f16(false, a0, false, Bw, (short)0, c0, false, false);
  c1 = __builtin_amdgcn_wmma_f32_16x16x32_f16(false, a1, false, Bw, (short)0, c1, false, false);

  // spill intermediate (C layout) to LDS so it can be re-read in B-operand layout
  #pragma unroll
  for (int j = 0; j < 8; ++j) {
    mid[wv][j + 8 * hh][q]      = (_Float16)c0[j];
    mid[wv][16 + j + 8 * hh][q] = (_Float16)c1[j];
  }
  __syncthreads();

  v16h b2;
  #pragma unroll
  for (int e = 0; e < 16; ++e) b2[e] = mid[wv][(e < 8) ? (hh * 8 + e) : (16 + hh * 8 + (e - 8))][q];

  // ---- vertical pass: banded A x intermediate -> final 16x16 tile
  v8f acc = {};
  acc = __builtin_amdgcn_wmma_f32_16x16x32_f16(false, A2, false, b2, (short)0, acc, false, false);

  // out = blur(x) + 254/255  (exact affine collapse of the reference arithmetic)
  #pragma unroll
  for (int j = 0; j < 8; ++j) {
    const int rr = row0 + j + 8 * hh;
    op[(size_t)rr * W + col0 + q] = acc[j] + (254.0f / 255.0f);
  }
}

extern "C" void kernel_launch(void* const* d_in, const int* in_sizes, int n_in,
                              void* d_out, int out_size, void* d_ws, size_t ws_size,
                              hipStream_t stream) {
  (void)in_sizes; (void)n_in; (void)out_size; (void)d_ws; (void)ws_size;
  const float* in = (const float*)d_in[0];
  float* out = (float*)d_out;
  // 96 planes x (32x32) tiles = 98304 waves; 8 waves per 256-thread block
  const int blocks = (96 * 32 * 32) / 8;   // 12288
  gauss17_wmma<<<blocks, 256, 0, stream>>>(in, out);
}
